// LSTMBlock_15771119911514
// MI455X (gfx1250) — matrix operand order repaired and checked by reading, not compiled, not run on hardware
//
#include <hip/hip_runtime.h>

// Problem constants (reference: B=64, T=512, I=128, H=256)
#define Bv 64
#define Tv 512
#define Iv 128
#define Hv 256
#define KC  384          // fused K = I + H
#define KSTEPS 12        // KC / 32 (bf16 WMMA K=32)
#define LDA 392          // padded row stride in ushorts (784B): rows land on distinct bank groups
#define NWG 8            // 2 directions x 4 hidden-slices
#define HS  64           // hidden units owned per workgroup

// LDS layout (dynamic): WL[256][392] bf16 weights slice, then A[64][392] bf16 [x_t | h]
#define WL_USHORTS (256 * LDA)           // 100,352 ushorts = 200,704 B
#define A_USHORTS  (Bv * LDA)            // 25,088 ushorts =  50,176 B
#define SMEM_BYTES ((WL_USHORTS + A_USHORTS) * 2)   // 250,880 B

// hard scheduling fence: keep prefetched ds_loads ahead of the consuming WMMAs
#define SCHED_FENCE() __builtin_amdgcn_sched_barrier(0)

typedef __attribute__((ext_vector_type(16))) __bf16 v16bf;
typedef __attribute__((ext_vector_type(8)))  float  v8f;

union BF16Frag { uint4 q[2]; v16bf v; };

__device__ __forceinline__ unsigned short f2bf(float f) {
  unsigned int u = __float_as_uint(f);
  u += 0x7FFFu + ((u >> 16) & 1u);   // round-to-nearest-even to bf16
  return (unsigned short)(u >> 16);
}
__device__ __forceinline__ float sigm(float x)      { return 1.0f / (1.0f + __expf(-x)); }
__device__ __forceinline__ float tanh_fast(float x) { return 2.0f / (1.0f + __expf(-2.0f * x)) - 1.0f; }

// CDNA5 async global->LDS DMA of one 16B chunk (ASYNCcnt-tracked)
__device__ __forceinline__ void async_cp16(const unsigned short* ldst, const unsigned short* gsrc) {
  unsigned int  lo = (unsigned int)(unsigned long long)ldst;   // low 32 bits of flat LDS ptr = LDS offset
  unsigned long long ga = (unsigned long long)gsrc;
  asm volatile("global_load_async_to_lds_b128 %0, %1, off" :: "v"(lo), "v"(ga) : "memory");
}
__device__ __forceinline__ void async_wait0() {
  asm volatile("s_wait_asynccnt 0" ::: "memory");
}

// ---- Phase 0a: x fp32 -> bf16 ([B][T][I]) ----
__global__ void conv_x(const float* __restrict__ x, unsigned short* __restrict__ xb, int n) {
  int idx = blockIdx.x * blockDim.x + threadIdx.x;
  if (idx < n) xb[idx] = f2bf(x[idx]);
}

// ---- Phase 0b: pack [w_ih | w_hh] -> bf16 [4H][384], fuse biases ----
__global__ void prep_dir(const float* __restrict__ wih, const float* __restrict__ whh,
                         const float* __restrict__ bih, const float* __restrict__ bhh,
                         unsigned short* __restrict__ wcat, float* __restrict__ bias) {
  int idx = blockIdx.x * blockDim.x + threadIdx.x;
  if (idx < 4 * Hv) bias[idx] = bih[idx] + bhh[idx];
  if (idx < 4 * Hv * KC) {
    int n = idx / KC, k = idx % KC;
    float v = (k < Iv) ? wih[n * Iv + k] : whh[n * Hv + (k - Iv)];
    wcat[idx] = f2bf(v);
  }
}

// ---- Phase 0c: zero h ping-pong buffers and per-step barrier counters ----
__global__ void init_sync(unsigned short* __restrict__ hbuf, int nh,
                          int* __restrict__ bar, int nb) {
  int i = blockIdx.x * blockDim.x + threadIdx.x;
  if (i < nh) hbuf[i] = 0;
  if (i < nb) bar[i] = 0;
}

// ---- Phase 1: persistent weights-stationary recurrence.
// grid = 8 WGs x 512 threads: blockIdx -> (dir = b>>2, hs = b&3).
// Weight slice (192KB) lives in LDS all 512 steps; h exchanged through L2
// (bf16, step-parity ping-pong) with a per-step release/acquire barrier
// (hardware s_cluster_barrier when dispatched as a cluster). LDS operand
// fetches are software-pipelined with sched_barrier fences so each k-step's
// ds_load latency hides behind the previous k-step's 4 WMMAs. ----
__launch_bounds__(512, 1)
__global__ void lstm_recurrence(const unsigned short* __restrict__ xbf,   // [B][T][I] bf16
                                const unsigned short* __restrict__ wcat,  // [2][4H][384] bf16
                                const float* __restrict__ bias,           // [2][4H]
                                unsigned short* __restrict__ hbuf,        // [2 buf][2 dir][64][256] bf16
                                int* __restrict__ bar,                    // [T]
                                float* __restrict__ out) {                // [B][T][2H] f32
  extern __shared__ __align__(16) unsigned short smem[];
  unsigned short* WL = smem;                 // [256][LDA]: row = gate*64 + local hidden col
  unsigned short* A  = smem + WL_USHORTS;    // [64][LDA]:  cols 0..127 = x_t, 128..383 = h_{t-1}

  const int dir = blockIdx.x >> 2;
  const int hs  = blockIdx.x & 3;            // hidden slice: global cols [hs*64, hs*64+64)
  const unsigned short* Wg = wcat + (size_t)dir * 4 * Hv * KC;
  const float* bs = bias + dir * 4 * Hv;

  const int tid  = threadIdx.x;
  const int lane = tid & 31;
  const int wave = tid >> 5;
  const int mt   = wave & 3;                 // batch tile
  const int hl   = wave >> 2;                // local hidden tile (0..3)
  const int half = lane >> 4;
  const int l15  = lane & 15;

  const bool in_cluster = __builtin_amdgcn_cluster_id_x() != 0;

  // ---- stage weight slice into LDS once (async DMA, 12288 x 16B chunks) ----
  for (int c = tid; c < 256 * 48; c += 512) {
    int r = c / 48, cc = c - r * 48;         // r = gate*64 + hloc
    int g = r >> 6, hloc = r & 63;
    const unsigned short* src = Wg + ((size_t)(g * Hv + hs * HS + hloc)) * KC + cc * 8;
    async_cp16(&WL[r * LDA + cc * 8], src);
  }

  float bias_r[4];
#pragma unroll
  for (int g = 0; g < 4; ++g)
    bias_r[g] = bs[g * Hv + hs * HS + hl * 16 + l15];

  float c_state[8];
#pragma unroll
  for (int v = 0; v < 8; ++v) c_state[v] = 0.0f;

  // stationary pieces of the per-lane fragment addresses
  const int arow = mt * 16 + l15;
  const unsigned short* Abase  = &A[arow * LDA + half * 8];                 // + k*32 (+16 for upper half)
  const unsigned short* WLbase = &WL[(hl * 16 + l15) * LDA + half * 8];     // + g*HS*LDA + k*32

  // prologue: stage x for step 0 (overlaps weight staging)
  {
    const int t0 = dir ? (Tv - 1) : 0;
    for (int c = tid; c < 1024; c += 512) {
      int row = c >> 4, cc = c & 15;
      async_cp16(&A[row * LDA + cc * 8], xbf + ((size_t)row * Tv + t0) * Iv + cc * 8);
    }
  }
  __syncthreads();

  for (int step = 0; step < Tv; ++step) {
    const int t  = dir ? (Tv - 1 - step) : step;
    const int rb = step & 1;                  // h read buffer (ping-pong)
    const int wb = rb ^ 1;                    // h write buffer

    // stage full h_{t-1} into A cols [128,384): 2048 x 16B chunks
    for (int c = tid; c < 2048; c += 512) {
      int row = c >> 5, cc = c & 31;
      async_cp16(&A[row * LDA + Iv + cc * 8],
                 hbuf + (((size_t)(rb * 2 + dir) * Bv + row) * Hv) + cc * 8);
    }
    async_wait0();                            // x slab (issued last step) + h slab complete
    __syncthreads();

    v8f acc[4];
#pragma unroll
    for (int g = 0; g < 4; ++g) {
      v8f z = {0.f, 0.f, 0.f, 0.f, 0.f, 0.f, 0.f, 0.f};
      acc[g] = z;
    }

    // ---- software-pipelined GEMM with hard scheduling fences ----
    BF16Frag af[2], bfr[2][4];
    {
      const unsigned short* ap = Abase;
      af[0].q[0] = *reinterpret_cast<const uint4*>(ap);
      af[0].q[1] = *reinterpret_cast<const uint4*>(ap + 16);
#pragma unroll
      for (int g = 0; g < 4; ++g) {
        const unsigned short* bp = WLbase + g * (HS * LDA);
        bfr[0][g].q[0] = *reinterpret_cast<const uint4*>(bp);
        bfr[0][g].q[1] = *reinterpret_cast<const uint4*>(bp + 16);
      }
    }
#pragma unroll
    for (int k = 0; k < KSTEPS; ++k) {
      const int cur = k & 1, nxt = cur ^ 1;
      if (k + 1 < KSTEPS) {
        const unsigned short* ap = Abase + (k + 1) * 32;
        af[nxt].q[0] = *reinterpret_cast<const uint4*>(ap);
        af[nxt].q[1] = *reinterpret_cast<const uint4*>(ap + 16);
#pragma unroll
        for (int g = 0; g < 4; ++g) {
          const unsigned short* bp = WLbase + g * (HS * LDA) + (k + 1) * 32;
          bfr[nxt][g].q[0] = *reinterpret_cast<const uint4*>(bp);
          bfr[nxt][g].q[1] = *reinterpret_cast<const uint4*>(bp + 16);
        }
      }
      SCHED_FENCE();   // k+1 prefetch must be issued before k's WMMAs
#pragma unroll
      for (int g = 0; g < 4; ++g)
        acc[g] = __builtin_amdgcn_wmma_f32_16x16x32_bf16(
            false, af[cur].v, false, bfr[cur][g].v, (short)0, acc[g], false, false);
    }

    __syncthreads();   // all waves done reading A -> safe to restage x region

    // stage x for step+1 now; overlaps nonlinearities + cross-WG barrier
    if (step + 1 < Tv) {
      const int tn = dir ? (Tv - 2 - step) : (step + 1);
      for (int c = tid; c < 1024; c += 512) {
        int row = c >> 4, cc = c & 15;
        async_cp16(&A[row * LDA + cc * 8], xbf + ((size_t)row * Tv + tn) * Iv + cc * 8);
      }
    }

    // Gates + cell update in C/D register layout:
    // lane l, slot v holds (m = mt*16 + v + 8*(l>=16), n = l&15).
    const int hcg = hs * HS + hl * 16 + l15;  // global hidden col
#pragma unroll
    for (int v = 0; v < 8; ++v) {
      float gi = acc[0][v] + bias_r[0];
      float gf = acc[1][v] + bias_r[1];
      float gg = acc[2][v] + bias_r[2];
      float go = acc[3][v] + bias_r[3];
      float c  = sigm(gf) * c_state[v] + sigm(gi) * tanh_fast(gg);
      c_state[v] = c;
      float h  = sigm(go) * tanh_fast(c);
      const int m = mt * 16 + v + half * 8;
      out[((size_t)m * Tv + t) * (2 * Hv) + dir * Hv + hcg] = h;            // fp32 output
      hbuf[(((size_t)(wb * 2 + dir) * Bv + m) * Hv) + hcg] = f2bf(h);       // h_t -> L2 exchange
    }

    // ---- cross-WG step barrier ----
    __threadfence();          // make h stores visible at device scope
    __syncthreads();
    if (in_cluster) {
      __builtin_amdgcn_s_cluster_barrier();   // hardware path when dispatched as a cluster
      __threadfence();
    } else {
      if (tid == 0)
        __hip_atomic_fetch_add(&bar[step], 1, __ATOMIC_RELEASE, __HIP_MEMORY_SCOPE_AGENT);
      while (__hip_atomic_load(&bar[step], __ATOMIC_ACQUIRE, __HIP_MEMORY_SCOPE_AGENT) < NWG)
        __builtin_amdgcn_s_sleep(2);
    }
    __syncthreads();
  }
}

extern "C" void kernel_launch(void* const* d_in, const int* in_sizes, int n_in,
                              void* d_out, int out_size, void* d_ws, size_t ws_size,
                              hipStream_t stream) {
  const float* x     = (const float*)d_in[0];
  const float* wih_f = (const float*)d_in[1];
  const float* whh_f = (const float*)d_in[2];
  const float* bih_f = (const float*)d_in[3];
  const float* bhh_f = (const float*)d_in[4];
  const float* wih_b = (const float*)d_in[5];
  const float* whh_b = (const float*)d_in[6];
  const float* bih_b = (const float*)d_in[7];
  const float* bhh_b = (const float*)d_in[8];
  float* out = (float*)d_out;

  // workspace layout (~10.1 MB)
  char* ws = (char*)d_ws;
  unsigned short* xbf  = (unsigned short*)ws;                          // 8,388,608 B
  unsigned short* wcat = (unsigned short*)(ws + 8388608);              // 1,572,864 B
  float*          bias = (float*)(ws + 9961472);                       //     8,192 B
  unsigned short* hbuf = (unsigned short*)(ws + 9969664);              //   131,072 B (2 buf x 2 dir x 64 x 256)
  int*            bar  = (int*)(ws + 10100736);                        //     2,048 B

  const int nx = Bv * Tv * Iv;            // 4,194,304
  conv_x<<<(nx + 255) / 256, 256, 0, stream>>>(x, xbf, nx);
  const int nw = 4 * Hv * KC;             // 393,216
  prep_dir<<<(nw + 255) / 256, 256, 0, stream>>>(wih_f, whh_f, bih_f, bhh_f, wcat, bias);
  prep_dir<<<(nw + 255) / 256, 256, 0, stream>>>(wih_b, whh_b, bih_b, bhh_b, wcat + nw, bias + 4 * Hv);
  const int nh = 2 * 2 * Bv * Hv;         // 65,536 ushorts
  init_sync<<<(nh + 255) / 256, 256, 0, stream>>>(hbuf, nh, bar, Tv);

  lstm_recurrence<<<NWG, 512, SMEM_BYTES, stream>>>(xbf, wcat, bias, hbuf, bar, out);
}
